// EERNN_62397284876772
// MI455X (gfx1250) — compile-verified, matrix operand order checked by reading, not run on hardware
//
#include <hip/hip_runtime.h>
#include <hip/hip_bf16.h>

#define BSZ   2
#define TLEN  50
#define PNUM  5111
#define PPAD  5120
#define UDIM  64
#define DIN   256   // 4U
#define KOUT  50
#define KPAD  64    // padded k-rows for branch-free WMMA stores

typedef __attribute__((ext_vector_type(16))) _Float16 v16h;
typedef __attribute__((ext_vector_type(8)))  float    v8f;

// ---------------------------------------------------------------------------
// Kernel 1: sequential LSTM (keras gate order i,f,g,o) + g = h @ W1[0:64,:]
// One workgroup of 256 threads; state + per-step activations in LDS.
// ---------------------------------------------------------------------------
__global__ __launch_bounds__(256) void eernn_lstm_kernel(
    const int* __restrict__ tgt, const int* __restrict__ cor,
    const float* __restrict__ X, const float* __restrict__ emb2,
    const float* __restrict__ Wk, const float* __restrict__ Wr,
    const float* __restrict__ bl, const float* __restrict__ W1,
    float* __restrict__ g_out)
{
    __shared__ float sxt[2][DIN];
    __shared__ float sz[2][DIN];
    __shared__ float sh[2][UDIM];
    __shared__ float sc[2][UDIM];
    __shared__ float hall[BSZ * TLEN][UDIM];   // 25.6 KB

    const int tid = threadIdx.x;
    if (tid < UDIM) {
        sh[0][tid] = 0.f; sh[1][tid] = 0.f;
        sc[0][tid] = 0.f; sc[1][tid] = 0.f;
    }
    __syncthreads();

    for (int t = 0; t < TLEN; ++t) {
        #pragma unroll
        for (int b = 0; b < 2; ++b) {
            const int tg = tgt[b * TLEN + t];
            const int cr = cor[b * TLEN + t];
            sxt[b][tid] = X[tg * (2 * UDIM) + (tid & 127)] * emb2[cr * DIN + tid];
        }
        __syncthreads();

        float z0 = bl[tid], z1 = bl[tid];
        for (int j = 0; j < DIN; ++j) {
            const float w = Wk[j * DIN + tid];
            z0 = fmaf(sxt[0][j], w, z0);
            z1 = fmaf(sxt[1][j], w, z1);
        }
        for (int j = 0; j < UDIM; ++j) {
            const float w = Wr[j * DIN + tid];
            z0 = fmaf(sh[0][j], w, z0);
            z1 = fmaf(sh[1][j], w, z1);
        }
        sz[0][tid] = z0; sz[1][tid] = z1;
        __syncthreads();

        if (tid < 2 * UDIM) {
            const int b = tid >> 6;
            const int n = tid & 63;
            const float zi = sz[b][n];
            const float zf = sz[b][UDIM + n];
            const float zg = sz[b][2 * UDIM + n];
            const float zo = sz[b][3 * UDIM + n];
            const float si = 1.f / (1.f + __expf(-zi));
            const float sf = 1.f / (1.f + __expf(-zf));
            const float so = 1.f / (1.f + __expf(-zo));
            const float c = sf * sc[b][n] + si * tanhf(zg);
            const float h = so * tanhf(c);
            sc[b][n] = c; sh[b][n] = h;
            hall[b * TLEN + t][n] = h;
        }
        __syncthreads();
    }

    for (int idx = tid; idx < BSZ * TLEN * KOUT; idx += 256) {
        const int bt = idx / KOUT;
        const int k  = idx - bt * KOUT;
        float acc = 0.f;
        for (int j = 0; j < UDIM; ++j)
            acc = fmaf(hall[bt][j], W1[j * KOUT + k], acc);
        g_out[idx] = acc;
    }
}

// ---------------------------------------------------------------------------
// Kernel 2: Xpart[k][p] = X[p,:] @ W1[64:192, k] + b1[k]   via WMMA f16->f32.
// B fragments pre-swizzled into LDS (fragment order), hoisted to registers and
// reused across 8 M-tiles. A rows use b128 vector loads through clamped
// (branch-free) addresses. Xpart is padded to 64 k-rows x PPAD so the store
// epilogue is fully unconditional (no exec-mask churn); padding is never read.
// ---------------------------------------------------------------------------
__global__ __launch_bounds__(128) void eernn_xpart_wmma_kernel(
    const float* __restrict__ X, const float* __restrict__ W1,
    const float* __restrict__ b1, float* __restrict__ Xpart)
{
    // [ntile][kt][lane][e] fragment-order B tiles, f16: 4*4*32*16 = 8192 halves
    __shared__ __align__(32) _Float16 sBf[8192];
    const int tid = threadIdx.x;
    const float* __restrict__ W1b = W1 + 64 * KOUT;   // rows 64..191 of W1

    for (int idx = tid; idx < 8192; idx += 128) {
        const int e    = idx & 15;
        const int lane = (idx >> 4) & 31;
        const int kt   = (idx >> 9) & 3;
        const int nt   = idx >> 11;
        const int hid  = lane >> 4;
        const int n    = nt * 16 + (lane & 15);
        const int k    = kt * 32 + hid * 8 + ((e < 8) ? e : (e + 8));
        const int nc   = (n < KOUT) ? n : (KOUT - 1);      // clamp, never stored
        sBf[idx] = (_Float16)W1b[k * KOUT + nc];
    }
    __syncthreads();

    const int wv      = tid >> 5;   // wave id == ntile (0..3)
    const int lane    = tid & 31;
    const int half_id = lane >> 4;
    const int lidx    = lane & 15;

    v16h bfrag[4];
    #pragma unroll
    for (int kt = 0; kt < 4; ++kt)
        bfrag[kt] = *(const v16h*)&sBf[((wv * 4 + kt) * 32 + lane) * 16];

    // Hoisted, branch-free bias load (column fixed per lane across all mtiles).
    const int   kcol = wv * 16 + lidx;
    const int   kcc  = (kcol < KOUT) ? kcol : (KOUT - 1);
    const float bias = b1[kcc];
    float* __restrict__ xout = Xpart + (size_t)kcol * PPAD;

    for (int mt = 0; mt < 8; ++mt) {
        const int mtile = blockIdx.x * 8 + mt;
        int m = mtile * 16 + lidx;
        m = (m < PNUM) ? m : (PNUM - 1);                   // clamp, never stored
        const float4* __restrict__ xrow = (const float4*)(X + (size_t)m * 128);

        v8f acc = {};
        #pragma unroll
        for (int kt = 0; kt < 4; ++kt) {
            const int q0 = kt * 8 + half_id * 2;       // (kt*32 + hid*8) / 4
            const int q1 = kt * 8 + 4 + half_id * 2;   // (kt*32 + 16 + hid*8) / 4
            const float4 x0 = xrow[q0], x1 = xrow[q0 + 1];
            const float4 x2 = xrow[q1], x3 = xrow[q1 + 1];
            v16h a;
            a[0]  = (_Float16)x0.x; a[1]  = (_Float16)x0.y;
            a[2]  = (_Float16)x0.z; a[3]  = (_Float16)x0.w;
            a[4]  = (_Float16)x1.x; a[5]  = (_Float16)x1.y;
            a[6]  = (_Float16)x1.z; a[7]  = (_Float16)x1.w;
            a[8]  = (_Float16)x2.x; a[9]  = (_Float16)x2.y;
            a[10] = (_Float16)x2.z; a[11] = (_Float16)x2.w;
            a[12] = (_Float16)x3.x; a[13] = (_Float16)x3.y;
            a[14] = (_Float16)x3.z; a[15] = (_Float16)x3.w;
            acc = __builtin_amdgcn_wmma_f32_16x16x32_f16(
                false, a, false, bfrag[kt], (short)0, acc, false, false);
        }

        // Branch-free epilogue: all 64 k-rows and all p < PPAD are legal in ws.
        const int pbase = mtile * 16 + half_id * 8;
        #pragma unroll
        for (int r = 0; r < 8; ++r)
            xout[pbase + r] = acc[r] + bias;
    }
}

// ---------------------------------------------------------------------------
// Kernel 3: reverse cumulative rank-1 attention + ReLU MLP head, fused.
// r[b,s,p] = relu(Cum[b,s,p,:] + Xpart[p,:]) . W2 + b2
// Cum[b,s] = Cum[b,s+1] + a[b,s,p] * g[b,s,:]   (tril => sum over t>=s)
// ---------------------------------------------------------------------------
__global__ __launch_bounds__(256) void eernn_head_kernel(
    const int* __restrict__ tgt, const float* __restrict__ cosX,
    const float* __restrict__ g, const float* __restrict__ Xpart,
    const float* __restrict__ W2, const float* __restrict__ b2,
    float* __restrict__ out)
{
    __shared__ float gsh[TLEN * KOUT];  // this batch's g, 10 KB
    __shared__ int   stgt[TLEN];
    __shared__ float sw2[KOUT];

    const int b   = blockIdx.y;
    const int tid = threadIdx.x;
    const int p   = blockIdx.x * blockDim.x + tid;

    for (int i = tid; i < TLEN * KOUT; i += 256) gsh[i] = g[b * TLEN * KOUT + i];
    if (tid < TLEN) stgt[tid] = tgt[b * TLEN + tid];
    if (tid < KOUT) sw2[tid] = W2[tid];
    __syncthreads();

    const bool pv = (p < PNUM);
    const int  pc = pv ? p : (PNUM - 1);   // clamped: loads branch-free
    float S[KOUT];
    float xw[KOUT];
    #pragma unroll
    for (int k = 0; k < KOUT; ++k) S[k] = 0.f;
    #pragma unroll
    for (int k = 0; k < KOUT; ++k) xw[k] = Xpart[k * PPAD + pc];
    const float b2v = b2[0];

    for (int s = TLEN - 1; s >= 0; --s) {
        if (s > 0)  // hide DRAM latency of the next gathered cos_X row
            __builtin_prefetch(&cosX[(size_t)stgt[s - 1] * PNUM + pc], 0, 1);
        const float av = cosX[(size_t)stgt[s] * PNUM + pc];
        const float* gs = &gsh[s * KOUT];
        float o = b2v;
        #pragma unroll
        for (int k = 0; k < KOUT; ++k) {
            S[k] = fmaf(av, gs[k], S[k]);
            o = fmaf(fmaxf(S[k] + xw[k], 0.f), sw2[k], o);
        }
        if (pv) out[(size_t)(b * TLEN + s) * PNUM + p] = o;
    }
}

// ---------------------------------------------------------------------------
extern "C" void kernel_launch(void* const* d_in, const int* in_sizes, int n_in,
                              void* d_out, int out_size, void* d_ws, size_t ws_size,
                              hipStream_t stream) {
    (void)in_sizes; (void)n_in; (void)out_size; (void)ws_size;
    const int*   tgt  = (const int*)d_in[0];
    const int*   cor  = (const int*)d_in[1];
    // d_in[2] = num_pro (scalar, unused), d_in[5] = trimatrix (structure hardcoded)
    const float* X    = (const float*)d_in[3];
    const float* cosX = (const float*)d_in[4];
    const float* emb2 = (const float*)d_in[6];
    const float* Wk   = (const float*)d_in[7];
    const float* Wr   = (const float*)d_in[8];
    const float* bl   = (const float*)d_in[9];
    const float* W1   = (const float*)d_in[10];
    const float* b1   = (const float*)d_in[11];
    const float* W2   = (const float*)d_in[12];
    const float* b2   = (const float*)d_in[13];
    float* outp = (float*)d_out;

    float* g     = (float*)d_ws;     // B*T*50 = 5000 floats (padded to 5120)
    float* Xpart = g + PPAD;         // KPAD(64) * PPAD floats, k-major, padded

    eernn_lstm_kernel<<<1, 256, 0, stream>>>(tgt, cor, X, emb2, Wk, Wr, bl, W1, g);
    // 320 M-tiles, 8 per block, 4 waves (= 4 N-tiles) per block -> 40 blocks
    eernn_xpart_wmma_kernel<<<40, 128, 0, stream>>>(X, W1, b1, Xpart);
    eernn_head_kernel<<<dim3((PPAD + 255) / 256, BSZ), 256, 0, stream>>>(
        tgt, cosX, g, Xpart, W2, b2, outp);
}
